// StencilConv3d_89258010345715
// MI455X (gfx1250) — matrix operand order repaired and checked by reading, not compiled, not run on hardware
//
#include <hip/hip_runtime.h>

// Problem constants (from reference)
#define B_    4
#define CIN   32
#define COUT  64
#define T_    8
#define DD    32
#define HH    64
#define WW    64
#define HWP   (HH*WW)       // 4096
#define VOL   (DD*HWP)      // 131072
#define NS    27            // 3x3x3 stencil taps

// Workgroup spatial tile: 4 x 8 x 8 = 256 points, 8 waves x 32 points
#define TZ 4
#define TY 8
#define TX 8
#define HZ (TZ+2)
#define HY (TY+2)
#define HX (TX+2)
#define NHP (HZ*HY*HX)      // 600 halo points
#define XPITCH 80           // bytes per halo point in LDS (64 used + 16 pad; keeps b128 16B-aligned, spreads banks)

#define WF_PER_B (NS*4*2*32*8)     // bf16 elements of fused weight per batch = 55296
#define WF_BYTES (WF_PER_B*2)      // 110592 B
#define XT_BYTES (NHP*XPITCH)      // 48000 B
#define SMEM_BYTES (XT_BYTES + WF_BYTES)

typedef __attribute__((ext_vector_type(16))) __bf16 v16bf;
typedef __attribute__((ext_vector_type(8)))  float  v8f;

union FragU { v16bf v; uint4 q[2]; };

// ---------------------------------------------------------------------------
// Prep: Wf[b,o,c,s] = sum_t weight[o,c,t] * stencils[b,t,s], emitted in bf16
// directly in B-fragment lane order:
//   flat = ((b*27+s)*2048) + (((nt*2+q)*32 + lane)*8 + e)
//   lane: kh = lane>>4 (K half), nn = lane&15 (column n)
//   element value = Wf[o = nt*16+nn][c = kh*16 + q*8 + e]
// ---------------------------------------------------------------------------
__global__ void wf_prep_kernel(const float* __restrict__ weight,
                               const float* __restrict__ sten,
                               __bf16* __restrict__ wf) {
  int i = blockIdx.x * 256 + threadIdx.x;      // 0 .. 221183
  int bs = i >> 11;                            // 2048 elements per (b,s)
  int b = bs / NS, s = bs % NS;
  int r = i & 2047;
  int nt   = r >> 9;
  int q    = (r >> 8) & 1;
  int lane = (r >> 3) & 31;
  int e    = r & 7;
  int c = ((lane >> 4) << 4) + q * 8 + e;
  int o = (nt << 4) + (lane & 15);
  float acc = 0.f;
#pragma unroll
  for (int t = 0; t < T_; ++t)
    acc += weight[(o * CIN + c) * T_ + t] * sten[(b * T_ + t) * NS + s];
  wf[i] = (__bf16)acc;
}

// ---------------------------------------------------------------------------
// Main: implicit-GEMM stencil conv.  One WG = 256 threads = (4,8,8) point tile.
// K loop over 27 stencil offsets, K-chunk = 32 channels = one bf16 WMMA K.
// ---------------------------------------------------------------------------
extern __shared__ char smem[];

__global__ void __launch_bounds__(256)
stencil_wmma_kernel(const float* __restrict__ x,
                    const __bf16* __restrict__ wf,
                    const float* __restrict__ bias,
                    float* __restrict__ out) {
  char* xlds = smem;                // XT_BYTES: bf16 x tile, [haloPoint][channel]
  char* wlds = smem + XT_BYTES;     // WF_BYTES: B fragments for this batch

  const int tid = threadIdx.x;
  const int blk = blockIdx.x;
  const int b   = blk >> 9;         // 512 tiles per batch
  const int trm = blk & 511;
  const int z0  = (trm >> 6) * TZ;
  const int y0  = ((trm >> 3) & 7) * TY;
  const int x0  = (trm & 7) * TX;

  // ---- stage fused weights (whole batch, 108 KB) into LDS ----
  {
    const uint4* src = reinterpret_cast<const uint4*>(wf + (size_t)b * WF_PER_B);
    uint4* dst = reinterpret_cast<uint4*>(wlds);
#pragma unroll
    for (int i = tid; i < WF_BYTES / 16; i += 256) dst[i] = src[i];
  }

  // ---- stage x tile + halo into LDS as bf16, channel-contiguous per point ----
  for (int i = tid; i < NHP * (CIN / 4); i += 256) {
    int cq = i / NHP;               // channel quad
    int hp = i % NHP;               // halo point (consecutive across lanes)
    int hz = hp / (HY * HX);
    int hr = hp % (HY * HX);
    int hy = hr / HX;
    int hx = hr % HX;
    int gz = z0 + hz - 1, gy = y0 + hy - 1, gx = x0 + hx - 1;
    bool in = ((unsigned)gz < DD) & ((unsigned)gy < HH) & ((unsigned)gx < WW);
    int c0 = cq * 4;
    float f0 = 0.f, f1 = 0.f, f2 = 0.f, f3 = 0.f;
    if (in) {
      size_t base = ((size_t)(b * CIN + c0)) * VOL + (size_t)gz * HWP + gy * WW + gx;
      f0 = x[base];
      f1 = x[base + VOL];
      f2 = x[base + 2 * (size_t)VOL];
      f3 = x[base + 3 * (size_t)VOL];
    }
    union { __bf16 h[4]; uint2 u; } p;
    p.h[0] = (__bf16)f0; p.h[1] = (__bf16)f1; p.h[2] = (__bf16)f2; p.h[3] = (__bf16)f3;
    *reinterpret_cast<uint2*>(xlds + hp * XPITCH + c0 * 2) = p.u;
  }
  __syncthreads();

  const int lane = tid & 31;
  const int wv   = tid >> 5;
  const int h    = lane >> 4;       // K-half for A fragments
  const int n15  = lane & 15;       // A row / C column within tile

  // Halo-point base index (stencil offset 0) for each of this wave's 2 M tiles
  int hpb[2];
#pragma unroll
  for (int mt = 0; mt < 2; ++mt) {
    int pidx = wv * 32 + mt * 16 + n15;
    int px = pidx & 7, py = (pidx >> 3) & 7, pz = pidx >> 6;
    hpb[mt] = pz * (HY * HX) + py * HX + px;
  }

  v8f acc[2][4] = {};

  for (int s = 0; s < NS; ++s) {
    int dz = s / 9, dy = (s / 3) % 3, dx = s % 3;
    int sOff = dz * (HY * HX) + dy * HX + dx;

    // B fragments for the 4 Cout tiles (from LDS, 2x b128 each)
    FragU bf[4];
    const char* wbase = wlds + s * 4096;
#pragma unroll
    for (int nt = 0; nt < 4; ++nt) {
      const char* p = wbase + nt * 1024;
      bf[nt].q[0] = *reinterpret_cast<const uint4*>(p + lane * 16);
      bf[nt].q[1] = *reinterpret_cast<const uint4*>(p + 512 + lane * 16);
    }

#pragma unroll
    for (int mt = 0; mt < 2; ++mt) {
      // A fragment: row = point, K = 32 channels (2x b128 from LDS)
      FragU af;
      const char* p = xlds + (hpb[mt] + sOff) * XPITCH + h * 32;
      af.q[0] = *reinterpret_cast<const uint4*>(p);
      af.q[1] = *reinterpret_cast<const uint4*>(p + 16);
#pragma unroll
      for (int nt = 0; nt < 4; ++nt) {
        acc[mt][nt] = __builtin_amdgcn_wmma_f32_16x16x32_bf16(
            false, af.v, false, bf[nt].v, (short)0, acc[mt][nt], false, false);
      }
    }
  }

  // ---- epilogue: + bias, store.  Each lane writes 2x float4 (contiguous W). ----
  float bv[4];
#pragma unroll
  for (int nt = 0; nt < 4; ++nt) bv[nt] = bias[nt * 16 + n15];

#pragma unroll
  for (int mt = 0; mt < 2; ++mt) {
    int pbase = wv * 32 + mt * 16 + 8 * h;   // + r gives point; x-coord == r
    int py = (pbase >> 3) & 7, pz = pbase >> 6;
    int gz = z0 + pz, gy = y0 + py;
#pragma unroll
    for (int nt = 0; nt < 4; ++nt) {
      int o = nt * 16 + n15;
      size_t base = ((size_t)(b * COUT + o)) * VOL + (size_t)gz * HWP + gy * WW + x0;
      float4 v0 = { acc[mt][nt][0] + bv[nt], acc[mt][nt][1] + bv[nt],
                    acc[mt][nt][2] + bv[nt], acc[mt][nt][3] + bv[nt] };
      float4 v1 = { acc[mt][nt][4] + bv[nt], acc[mt][nt][5] + bv[nt],
                    acc[mt][nt][6] + bv[nt], acc[mt][nt][7] + bv[nt] };
      *reinterpret_cast<float4*>(out + base)     = v0;
      *reinterpret_cast<float4*>(out + base + 4) = v1;
    }
  }
}

// ---------------------------------------------------------------------------
extern "C" void kernel_launch(void* const* d_in, const int* in_sizes, int n_in,
                              void* d_out, int out_size, void* d_ws, size_t ws_size,
                              hipStream_t stream) {
  const float* x      = (const float*)d_in[0];  // (4,32,32,64,64)
  const float* sten   = (const float*)d_in[1];  // (4,8,3,3,3)
  const float* weight = (const float*)d_in[2];  // (64,32,8)
  const float* bias   = (const float*)d_in[3];  // (64,)
  float* out          = (float*)d_out;          // (4,64,32,64,64)
  __bf16* wfrag       = (__bf16*)d_ws;          // 221184 bf16 = 442368 B

  // Fused-weight fragments: 221184 elements, 864 blocks x 256 threads exactly.
  wf_prep_kernel<<<864, 256, 0, stream>>>(weight, sten, wfrag);

  // 2048 tiles (4 batches x 512), 155 KB dynamic LDS each (CDNA5: 320 KB/WGP).
  stencil_wmma_kernel<<<2048, 256, SMEM_BYTES, stream>>>(x, wfrag, bias, out);
}
